// MultiHeadAttention_2499670966679
// MI455X (gfx1250) — matrix operand order repaired and checked by reading, not compiled, not run on hardware
//
#include <hip/hip_runtime.h>

typedef __attribute__((ext_vector_type(16))) _Float16 v16h;
typedef __attribute__((ext_vector_type(8)))  float    v8f;

#define BSZ   2
#define SLEN  2048
#define DM    1024
#define NHEAD 16
#define DHEAD 64

// fused-attention LDS layout (floats)
#define STRIP_LD    2052                       // 2048 + 4 pad: breaks bank alignment
#define SMEM_FLOATS (16 * STRIP_LD + 256 + 1024)
#define SMEM_BYTES  (SMEM_FLOATS * 4)

// ---------------- WMMA helpers (CDNA5 wave32, 16x16x32 f16 -> f32) ----------

static __device__ __forceinline__ v8f wmma_f16(v16h a, v16h b, v8f c) {
  // 8 args: (neg_a, A, neg_b, B, c_mod, C, reuse_a, reuse_b)
  return __builtin_amdgcn_wmma_f32_16x16x32_f16(false, a, false, b, (short)0, c,
                                                false, false);
}

// A fragment (16x32, f16): lane = (khalf<<4)|m ; element e:
//   K = k0 + 16*khalf + e   -> one contiguous 16-elem run
static __device__ __forceinline__ v16h load_a_f32(const float* __restrict__ A,
                                                  int lda, int m, int k0, int khalf) {
  const float* row = A + (size_t)m * lda + k0 + khalf * 16;
  v16h r;
#pragma unroll
  for (int i = 0; i < 8; ++i) r[i] = (_Float16)row[i];
#pragma unroll
  for (int i = 0; i < 8; ++i) r[8 + i] = (_Float16)row[8 + i];
  return r;
}

static __device__ __forceinline__ v16h load_a_f16(const _Float16* __restrict__ A,
                                                  int lda, int m, int k0, int khalf) {
  const _Float16* row = A + (size_t)m * lda + k0 + khalf * 16;
  v16h r;
#pragma unroll
  for (int i = 0; i < 8; ++i) r[i] = row[i];
#pragma unroll
  for (int i = 0; i < 8; ++i) r[8 + i] = row[8 + i];
  return r;
}

// A fragment sourced from an LDS f32 strip (emits ds_load + cvt)
static __device__ __forceinline__ v16h load_a_lds(const float* S, int ld, int m,
                                                  int k0, int khalf) {
  const float* row = S + m * ld + k0 + khalf * 16;
  v16h r;
#pragma unroll
  for (int i = 0; i < 8; ++i) r[i] = (_Float16)row[i];
#pragma unroll
  for (int i = 0; i < 8; ++i) r[8 + i] = (_Float16)row[8 + i];
  return r;
}

// B fragment (32x16) from row-major B[k][n]: lane = (khalf<<4)|n,
//   element e -> K = k0 + 16*khalf + e (coalesced across lanes)
static __device__ __forceinline__ v16h load_b_rm_f32(const float* __restrict__ B,
                                                     int ldb, int k0, int n, int khalf) {
  const float* p = B + (size_t)(k0 + 16 * khalf) * ldb + n;
  v16h r;
#pragma unroll
  for (int e = 0; e < 16; ++e) r[e] = (_Float16)p[(size_t)e * ldb];
  return r;
}

static __device__ __forceinline__ v16h load_b_rm_f16(const _Float16* __restrict__ B,
                                                     int ldb, int k0, int n, int khalf) {
  const _Float16* p = B + (size_t)(k0 + 16 * khalf) * ldb + n;
  v16h r;
#pragma unroll
  for (int e = 0; e < 16; ++e) r[e] = p[(size_t)e * ldb];
  return r;
}

// B fragment where logical B = X^T (column n of B is row n of X): contiguous K
static __device__ __forceinline__ v16h load_bt_f16(const _Float16* __restrict__ X,
                                                   int ldx, int n, int k0, int khalf) {
  const _Float16* p = X + (size_t)n * ldx + k0 + 16 * khalf;
  v16h r;
#pragma unroll
  for (int e = 0; e < 16; ++e) r[e] = p[e];
  return r;
}

// D/C 16x16 f32: lane = N + 16*(M>>3); VGPR v holds row M = v + 8*(lane>>4)
static __device__ __forceinline__ void store_d_f16(_Float16* __restrict__ C, int ldc,
                                                   int m0, int n0, v8f d, int lane) {
  int n = n0 + (lane & 15);
  int mb = m0 + (lane >> 4) * 8;
#pragma unroll
  for (int v = 0; v < 8; ++v) C[(size_t)(mb + v) * ldc + n] = (_Float16)d[v];
}

static __device__ __forceinline__ void store_d_lds_scaled(float* C, int ldc,
                                                          int m0, int n0, v8f d,
                                                          float s, int lane) {
  int n = n0 + (lane & 15);
  int mb = m0 + (lane >> 4) * 8;
#pragma unroll
  for (int v = 0; v < 8; ++v) C[(mb + v) * ldc + n] = d[v] * s;
}

// ---------------- Kernels ---------------------------------------------------

// C_f16[M,N] = A_f32[M,K] * W_f32[K,N]; M=BSZ*SLEN, K=N=DM
__global__ void mha_proj_gemm(const float* __restrict__ A, const float* __restrict__ W,
                              _Float16* __restrict__ C) {
  const int lane = threadIdx.x & 31, wave = threadIdx.x >> 5;
  const int khalf = lane >> 4, ml = lane & 15;
  const int m0 = blockIdx.y * 64 + (wave >> 1) * 32;
  const int n0 = blockIdx.x * 64 + (wave & 1) * 32;
  v8f acc[2][2] = {};
  for (int k0 = 0; k0 < DM; k0 += 32) {
    if (k0 + 32 < DM) {  // gfx1250 global_prefetch_b8 of next A tile (near scope)
      __builtin_prefetch(A + (size_t)(m0 + ml) * DM + k0 + 32, 0, 3);
      __builtin_prefetch(A + (size_t)(m0 + 16 + ml) * DM + k0 + 32, 0, 3);
    }
    v16h a0 = load_a_f32(A, DM, m0 + ml, k0, khalf);
    v16h a1 = load_a_f32(A, DM, m0 + 16 + ml, k0, khalf);
    v16h b0 = load_b_rm_f32(W, DM, k0, n0 + ml, khalf);
    v16h b1 = load_b_rm_f32(W, DM, k0, n0 + 16 + ml, khalf);
    acc[0][0] = wmma_f16(a0, b0, acc[0][0]);
    acc[0][1] = wmma_f16(a0, b1, acc[0][1]);
    acc[1][0] = wmma_f16(a1, b0, acc[1][0]);
    acc[1][1] = wmma_f16(a1, b1, acc[1][1]);
  }
#pragma unroll
  for (int i = 0; i < 2; ++i)
#pragma unroll
    for (int j = 0; j < 2; ++j)
      store_d_f16(C, DM, m0 + 16 * i, n0 + 16 * j, acc[i][j], lane);
}

// Fused scores + softmax + attn@V for a 16-query-row strip of one (b,h).
// The 16x2048 probability strip lives entirely in the 320KB WGP LDS:
// global attn traffic = one 537MB write instead of write+2x read+rewrite.
__global__ void mha_attn_fused(const _Float16* __restrict__ Qp,
                               const _Float16* __restrict__ Kp,
                               const _Float16* __restrict__ Vp,
                               float* __restrict__ attn,
                               _Float16* __restrict__ Oh) {
  extern __shared__ float smem[];
  float* strip   = smem;                    // 16 x STRIP_LD f32 (P strip)
  float* red     = smem + 16 * STRIP_LD;    // 256 f32 (softmax scratch + inv[16])
  float* partial = red + 256;               // 4 tiles x 256 f32 (attn@V partials)

  const int t = threadIdx.x;
  const int lane = t & 31, wave = t >> 5;
  const int khalf = lane >> 4, ml = lane & 15;
  const int bh = blockIdx.y, b = bh >> 4, h = bh & 15;
  const int m0g = blockIdx.x * 16;  // global query-row base of this strip

  const _Float16* Aq = Qp + (size_t)b * SLEN * DM + h * DHEAD;
  const _Float16* Ak = Kp + (size_t)b * SLEN * DM + h * DHEAD;
  const _Float16* Bv = Vp + (size_t)b * SLEN * DM + h * DHEAD;
  float* attnS = attn + (size_t)bh * SLEN * SLEN + (size_t)m0g * SLEN;

  // ---- Step 1: S strip (16 x 2048) -> LDS, wave w covers lk in [w*256,w*256+256)
  {
    v16h a0 = load_a_f16(Aq, DM, m0g + ml, 0, khalf);
    v16h a1 = load_a_f16(Aq, DM, m0g + ml, 32, khalf);
#pragma unroll
    for (int tile = 0; tile < 8; ++tile) {
      const int n0 = wave * 256 + tile * 32;
      v16h b00 = load_bt_f16(Ak, DM, n0 + ml, 0, khalf);
      v16h b01 = load_bt_f16(Ak, DM, n0 + ml, 32, khalf);
      v16h b10 = load_bt_f16(Ak, DM, n0 + 16 + ml, 0, khalf);
      v16h b11 = load_bt_f16(Ak, DM, n0 + 16 + ml, 32, khalf);
      v8f acc0 = {}, acc1 = {};
      acc0 = wmma_f16(a0, b00, acc0);
      acc0 = wmma_f16(a1, b01, acc0);
      acc1 = wmma_f16(a0, b10, acc1);
      acc1 = wmma_f16(a1, b11, acc1);
      store_d_lds_scaled(strip, STRIP_LD, 0, n0, acc0, 0.125f, lane);
      store_d_lds_scaled(strip, STRIP_LD, 0, n0 + 16, acc1, 0.125f, lane);
    }
  }
  __syncthreads();

  // ---- Step 2: row softmax in LDS; 16 threads per row, 128 elems each
  {
    const int r = t >> 4, seg = t & 15;
    float* rowp = strip + r * STRIP_LD + seg * 128;
    float lmax = rowp[0];
#pragma unroll 4
    for (int i = 1; i < 128; ++i) lmax = fmaxf(lmax, rowp[i]);
    red[r * 16 + seg] = lmax;
    __syncthreads();
    float rmax = red[r * 16];
#pragma unroll
    for (int jj = 1; jj < 16; ++jj) rmax = fmaxf(rmax, red[r * 16 + jj]);
    __syncthreads();
    float lsum = 0.0f;
#pragma unroll 4
    for (int i = 0; i < 128; ++i) {
      float p = expf(rowp[i] - rmax);
      rowp[i] = p;
      lsum += p;
    }
    red[r * 16 + seg] = lsum;
    __syncthreads();
    float rsum = 0.0f;
#pragma unroll
    for (int jj = 0; jj < 16; ++jj) rsum += red[r * 16 + jj];
    const float inv = 1.0f / rsum;
    __syncthreads();
    if (seg == 0) red[r] = inv;  // red[0..15] = per-row inverse sums
    __syncthreads();
    // normalize + single coalesced write of attn to HBM
    for (int idx = t; idx < 16 * SLEN; idx += 256) {
      const int rr = idx >> 11, cc = idx & (SLEN - 1);
      const float p = strip[rr * STRIP_LD + cc] * red[rr];
      strip[rr * STRIP_LD + cc] = p;
      attnS[(size_t)rr * SLEN + cc] = p;
    }
  }
  __syncthreads();

  // ---- Step 3: O strip (16 x 64) = P(16x2048, LDS) @ V(2048x64)
  // wave w: n-tile j = w&3, K partition (w>>2)*1024 .. +1024; LDS partial reduce
  {
    const int j = wave & 3;
    const int kbase = (wave >> 2) * 1024;
    v8f acc = {};
    for (int k0 = kbase; k0 < kbase + 1024; k0 += 32) {
      v16h a = load_a_lds(strip, STRIP_LD, ml, k0, khalf);
      v16h bf = load_b_rm_f16(Bv, DM, k0, j * 16 + ml, khalf);
      acc = wmma_f16(a, bf, acc);
    }
    if (wave >= 4) {
#pragma unroll
      for (int v = 0; v < 8; ++v) partial[j * 256 + lane * 8 + v] = acc[v];
    }
    __syncthreads();
    if (wave < 4) {
#pragma unroll
      for (int v = 0; v < 8; ++v) acc[v] += partial[j * 256 + lane * 8 + v];
      _Float16* Co = Oh + (size_t)b * SLEN * DM + h * DHEAD;
      store_d_f16(Co, DM, m0g, j * 16, acc, lane);
    }
  }
}

// out[m,n] = sum_k Oh[m,k]*Wo[k,n] + residual[m,n]
__global__ void mha_out_proj(const _Float16* __restrict__ Oh,
                             const float* __restrict__ Wo,
                             const float* __restrict__ resid,
                             float* __restrict__ out) {
  const int lane = threadIdx.x & 31, wave = threadIdx.x >> 5;
  const int khalf = lane >> 4, ml = lane & 15;
  const int m0 = blockIdx.y * 64 + (wave >> 1) * 32;
  const int n0 = blockIdx.x * 64 + (wave & 1) * 32;
  v8f acc[2][2] = {};
  for (int k0 = 0; k0 < DM; k0 += 32) {
    v16h a0 = load_a_f16(Oh, DM, m0 + ml, k0, khalf);
    v16h a1 = load_a_f16(Oh, DM, m0 + 16 + ml, k0, khalf);
    v16h b0 = load_b_rm_f32(Wo, DM, k0, n0 + ml, khalf);
    v16h b1 = load_b_rm_f32(Wo, DM, k0, n0 + 16 + ml, khalf);
    acc[0][0] = wmma_f16(a0, b0, acc[0][0]);
    acc[0][1] = wmma_f16(a0, b1, acc[0][1]);
    acc[1][0] = wmma_f16(a1, b0, acc[1][0]);
    acc[1][1] = wmma_f16(a1, b1, acc[1][1]);
  }
  const int n = lane & 15, mb = (lane >> 4) * 8;
#pragma unroll
  for (int i = 0; i < 2; ++i)
#pragma unroll
    for (int j = 0; j < 2; ++j) {
      const int mm = m0 + 16 * i + mb, nn = n0 + 16 * j + n;
#pragma unroll
      for (int v = 0; v < 8; ++v) {
        const size_t idx = (size_t)(mm + v) * DM + nn;
        out[idx] = acc[i][j][v] + resid[idx];
      }
    }
}

// ---------------- Host launcher --------------------------------------------

extern "C" void kernel_launch(void* const* d_in, const int* in_sizes, int n_in,
                              void* d_out, int out_size, void* d_ws, size_t ws_size,
                              hipStream_t stream) {
  (void)in_sizes; (void)n_in; (void)out_size; (void)ws_size;
  const float* q  = (const float*)d_in[0];
  const float* k  = (const float*)d_in[1];
  const float* v  = (const float*)d_in[2];
  const float* Wq = (const float*)d_in[3];
  const float* Wk = (const float*)d_in[4];
  const float* Wv = (const float*)d_in[5];
  const float* Wo = (const float*)d_in[6];

  const size_t proj_elems = (size_t)BSZ * SLEN * DM;  // 4096*1024
  _Float16* Qp = (_Float16*)d_ws;
  _Float16* Kp = Qp + proj_elems;
  _Float16* Vp = Kp + proj_elems;
  _Float16* Oh = Vp + proj_elems;

  float* out  = (float*)d_out;
  float* attn = out + proj_elems;  // attn region: BS*NH*SLEN*SLEN f32

  (void)hipFuncSetAttribute((const void*)mha_attn_fused,
                            hipFuncAttributeMaxDynamicSharedMemorySize, SMEM_BYTES);

  const dim3 blk(128);                          // 4 waves of 32
  const dim3 gproj(DM / 64, (BSZ * SLEN) / 64); // 16 x 64

  mha_proj_gemm<<<gproj, blk, 0, stream>>>(q, Wq, Qp);
  mha_proj_gemm<<<gproj, blk, 0, stream>>>(k, Wk, Kp);
  mha_proj_gemm<<<gproj, blk, 0, stream>>>(v, Wv, Vp);

  // fused attention: 128 row-strips x 32 (b,h) pairs, 8 waves/block, 133KB LDS
  mha_attn_fused<<<dim3(SLEN / 16, BSZ * NHEAD), dim3(256), SMEM_BYTES, stream>>>(
      Qp, Kp, Vp, attn, Oh);

  mha_out_proj<<<gproj, blk, 0, stream>>>(Oh, Wo, q, out);
}
